// dcn_layer_28638841929772
// MI455X (gfx1250) — compile-verified, matrix operand order checked by reading, not run on hardware
//
#include <hip/hip_runtime.h>
#include <math.h>

// ---------------------------------------------------------------------------
// DCNv2 fused pipeline for gfx1250 (MI455X), bf16 WMMA 16x16x32, f32 accum.
// Shapes: B=8, C=64, H=W=64, O=64, K=3 (KK=9), N = B*H*W = 32768.
// ---------------------------------------------------------------------------

#define NPIX   32768      // B*H*W

typedef __bf16 v16bf __attribute__((ext_vector_type(16)));
typedef float  v8f   __attribute__((ext_vector_type(8)));
typedef unsigned int u32x4 __attribute__((ext_vector_type(4)));
typedef unsigned int u32x8 __attribute__((ext_vector_type(8)));

union BFrag {
    unsigned short us[16];
    int4           i4[2];
    v16bf          v;
};

__device__ __forceinline__ unsigned short f2bf(float f) {
    // round-to-nearest-even fp32 -> bf16
    unsigned int u = __float_as_uint(f);
    u += 0x7FFFu + ((u >> 16) & 1u);
    return (unsigned short)(u >> 16);
}

__device__ __forceinline__ v8f wmma_bf16(v16bf a, v16bf b, v8f c) {
    // D = A(16x32 bf16) x B(32x16 bf16) + C(16x16 f32)
    return __builtin_amdgcn_wmma_f32_16x16x32_bf16(
        /*neg_a=*/false, a, /*neg_b=*/false, b,
        /*c_mod=*/(short)0, c, /*reuse_a=*/false, /*reuse_b=*/false);
}

// TDM: DMA one 73728-byte fragment pack (9216 x 8B elements, single row)
// from global memory into LDS offset 0.  2-group 2D descriptor:
//   group0 = {ctrl(count=1), lds_addr=0, gaddr_lo, gaddr_hi | type=2}
//   group1: data_size=3(8B) @[17:16]; tensor_dim0=9216 @[79:48];
//           tensor_dim1=1 @[111:80]; tile_dim0=9216 @[127:112];
//           tile_dim1=1 @[143:128]; tensor_dim0_stride=9216 @[207:160]
__device__ __forceinline__ void tdm_load_pack_to_lds0(const void* gsrc) {
    const unsigned long long ga = (unsigned long long)gsrc;
    u32x4 g0 = { 1u, 0u, (unsigned)ga, (unsigned)(ga >> 32) | 0x80000000u };
    u32x8 g1 = { 0x00030000u, 0x24000000u, 0x00010000u, 0x24000000u,
                 1u, 9216u, 0u, 0u };
    asm volatile("tensor_load_to_lds %0, %1" :: "s"(g0), "s"(g1) : "memory");
    __builtin_amdgcn_s_wait_tensorcnt(0);
}

// A/B fragment element->K mapping (ISA 7.12.2, 16-bit, wave32):
//   lane group g = lane>>4;  element e (0..15):  k = (e<8 ? e : e+8) + g*8
// => elements 0..7  are K contiguous run [g*8,  g*8+8)
//    elements 8..15 are K contiguous run [16+g*8, 16+g*8+8)
// Inverse (k -> g,e):  g=(k>>3)&1 ; e=(k&7) | ((k>>4)<<3)

// ---------------------------------------------------------------------------
// Kernel 0a: pack om_weight [27][128][3][3] into A-fragments.
// K-order: k = cb*288 + tap*32 + ci_local.  36 K-steps, 2 M-tiles (rows 27..31 = 0).
// Layout: a1[((step*2 + mt)*32 + lane)*16 + e]
// ---------------------------------------------------------------------------
__global__ void __launch_bounds__(256) pack_a1_kernel(
    const float* __restrict__ om_w, unsigned short* __restrict__ a1)
{
    const int t    = blockIdx.x * 256 + threadIdx.x;   // 2304 threads
    const int lane = t & 31;
    const int mt   = (t >> 5) & 1;
    const int step = t >> 6;                           // 0..35
    const int g    = lane >> 4;
    const int m    = mt * 16 + (lane & 15);
    const int cb   = step / 9, tap = step % 9;
    unsigned short* dst = a1 + t * 16;
#pragma unroll
    for (int e = 0; e < 16; ++e) {
        const int kl = (e < 8 ? e : e + 8) + g * 8;    // 0..31
        const int ci = cb * 32 + kl;                   // concat channel
        const float v = (m < 27) ? om_w[(m * 128 + ci) * 9 + tap] : 0.0f;
        dst[e] = f2bf(v);
    }
}

// ---------------------------------------------------------------------------
// Kernel 0b: pack weight [64][64][3][3] into A-fragments.
// K-order: k2 = tap*64 + c -> s = tap*2 + (c>>5).  18 K-steps, 4 M-tiles.
// Layout: a2[((s*4 + mt)*32 + lane)*16 + e]
// ---------------------------------------------------------------------------
__global__ void __launch_bounds__(256) pack_a2_kernel(
    const float* __restrict__ wgt, unsigned short* __restrict__ a2)
{
    const int t    = blockIdx.x * 256 + threadIdx.x;   // 2304 threads
    const int lane = t & 31;
    const int mt   = (t >> 5) & 3;
    const int s    = t >> 7;                           // 0..17
    const int g    = lane >> 4;
    const int o    = mt * 16 + (lane & 15);
    const int tap  = s >> 1;
    unsigned short* dst = a2 + t * 16;
#pragma unroll
    for (int e = 0; e < 16; ++e) {
        const int kl = (e < 8 ? e : e + 8) + g * 8;
        const int c  = (s & 1) * 32 + kl;
        dst[e] = f2bf(wgt[(o * 64 + c) * 9 + tap]);
    }
}

// ---------------------------------------------------------------------------
// Kernel 1: offset/mask conv.  GEMM M=32(27), K=1152, N=32768, implicit im2col.
// One block per image row (b,h): 128 threads = 4 waves x 16 pixels.
// LDS: [0, 73728)   A1 fragment pack, DMA'd once by the TDM
//      [73728, ...) halo tile, channel-contiguous (stride 40 = 20 dwords,
//                   bank-conflict-free) -> B-fragment = 2x ds_load_b128.
// Staging: interior rows as float4 loads (row-level predicate only); the
// x=-1 / x=64 halo columns are always out of image -> zeroed once per block.
// Epilogue: + bias, sigmoid on channels 18..26, write om[27][NPIX].
// ---------------------------------------------------------------------------
__global__ void __launch_bounds__(128) om_conv_kernel(
    const float* __restrict__ feat, const float* __restrict__ degrad,
    const float* __restrict__ bias, const unsigned short* __restrict__ a1,
    float* __restrict__ om)
{
    extern __shared__ char smem[];
    int4* sAp = (int4*)smem;                                      // 73728 B
    unsigned short (*lds)[66][40] =
        (unsigned short (*)[66][40])(smem + 73728);               // [ry][x+1][ci]

    const int tid  = threadIdx.x;
    const int lane = tid & 31;
    const int wv   = tid >> 5;
    const int bh   = blockIdx.x;            // b*64 + h
    const int b    = bh >> 6, h = bh & 63;
    const int g    = lane >> 4;
    const int n    = wv * 16 + (lane & 15); // pixel (w) owned by this lane

    if (wv == 0) tdm_load_pack_to_lds0(a1);  // async DMA of A-fragments

    // halo columns (image x = -1 and x = 64) are always zero; write once
    for (int i = tid; i < 192; i += 128) {
        const int ci = i & 31;
        const int r2 = i >> 5;              // 0..5
        lds[r2 >> 1][(r2 & 1) * 65][ci] = 0;
    }

    v8f acc0 = {}, acc1 = {};

    for (int cb = 0; cb < 4; ++cb) {
        const float* src = (cb < 2) ? feat : degrad;
        const int ch0 = (cb & 1) * 32;
        // interior stage: 32 ch x 3 rows x 16 quads of 4 pixels (float4 loads)
        for (int i = tid; i < 32 * 3 * 16; i += 128) {
            const int ci   = i / 48;
            const int rest = i - ci * 48;
            const int ry   = rest >> 4;
            const int x0   = (rest & 15) << 2;
            const int y    = h - 1 + ry;
            float4 v = make_float4(0.f, 0.f, 0.f, 0.f);
            if ((unsigned)y < 64u)
                v = *(const float4*)(src + ((b * 64 + ch0 + ci) * 64 + y) * 64 + x0);
            unsigned short* d = &lds[ry][x0 + 1][ci];
            d[0]   = f2bf(v.x);
            d[40]  = f2bf(v.y);
            d[80]  = f2bf(v.z);
            d[120] = f2bf(v.w);
        }
        __syncthreads();   // publish staged tile (and, on cb==0, the TDM'd A pack)

        for (int tap = 0; tap < 9; ++tap) {
            const int ky = tap / 3, kx = tap % 3;
            const int step = cb * 9 + tap;
            // A fragments from LDS (2 M-tiles)
            const int4* ap = sAp + ((step * 2) * 32 + lane) * 2;
            BFrag a0f, a1f, bf;
            a0f.i4[0] = ap[0];  a0f.i4[1] = ap[1];
            a1f.i4[0] = ap[64]; a1f.i4[1] = ap[65];
            // B fragment: two contiguous 8-channel runs -> 2x ds_load_b128
            const unsigned short* row = &lds[ky][n + kx][0];
            bf.i4[0] = *(const int4*)(row + g * 8);
            bf.i4[1] = *(const int4*)(row + 16 + g * 8);
            acc0 = wmma_bf16(a0f.v, bf.v, acc0);
            acc1 = wmma_bf16(a1f.v, bf.v, acc1);
        }
        __syncthreads();   // before re-staging the halo tile
    }

    const int pix = bh * 64 + n;            // b*4096 + h*64 + w
#pragma unroll
    for (int mt = 0; mt < 2; ++mt) {
        v8f acc = mt ? acc1 : acc0;
#pragma unroll
        for (int r = 0; r < 8; ++r) {
            const int ch = mt * 16 + g * 8 + r;    // C/D layout: M = g*8 + r
            if (ch < 27) {
                float v = acc[r] + bias[ch];
                if (ch >= 18) v = 1.0f / (1.0f + __expf(-v));   // sigmoid mask
                om[ch * NPIX + pix] = v;
            }
        }
    }
}

// ---------------------------------------------------------------------------
// Kernel 2: bilinear sampling; one thread per (tap, pixel); loops 64 channels,
// two channels per iteration so each store is one packed 32-bit write into the
// WMMA B-fragment layout:
//   sp[(((s*2048 + nb)*32 + g*16 + (pix&15)) << 4) + e],  s = tap*2 + (c>>5)
// ---------------------------------------------------------------------------
__global__ void __launch_bounds__(256) sample_pack_kernel(
    const float* __restrict__ feat, const float* __restrict__ om,
    unsigned short* __restrict__ sp)
{
    const int idx   = blockIdx.x * 256 + threadIdx.x;  // 294912 = 9*NPIX
    const int pixel = idx & (NPIX - 1);
    const int tap   = idx >> 15;
    const int b = pixel >> 12, h = (pixel >> 6) & 63, w = pixel & 63;
    const int ky = tap / 3, kx = tap % 3;

    const float dy  = om[(2 * tap)     * NPIX + pixel];
    const float dx  = om[(2 * tap + 1) * NPIX + pixel];
    const float msk = om[(18 + tap)    * NPIX + pixel];

    const float y = (float)(h - 1 + ky) + dy;
    const float x = (float)(w - 1 + kx) + dx;
    const float y0f = floorf(y), x0f = floorf(x);
    const int y0 = (int)y0f, x0 = (int)x0f;
    const int y1 = y0 + 1,   x1 = x0 + 1;
    const float wy = y - y0f, wx = x - x0f;
    const bool vy0 = (unsigned)y0 < 64u, vy1 = (unsigned)y1 < 64u;
    const bool vx0 = (unsigned)x0 < 64u, vx1 = (unsigned)x1 < 64u;
    const int yc0 = min(max(y0, 0), 63), yc1 = min(max(y1, 0), 63);
    const int xc0 = min(max(x0, 0), 63), xc1 = min(max(x1, 0), 63);
    const int i00 = yc0 * 64 + xc0, i01 = yc0 * 64 + xc1;
    const int i10 = yc1 * 64 + xc0, i11 = yc1 * 64 + xc1;
    // fold mask + validity into the corner weights
    const float w00 = (vy0 && vx0) ? (1.f - wy) * (1.f - wx) * msk : 0.f;
    const float w01 = (vy0 && vx1) ? (1.f - wy) * wx         * msk : 0.f;
    const float w10 = (vy1 && vx0) ? wy         * (1.f - wx) * msk : 0.f;
    const float w11 = (vy1 && vx1) ? wy         * wx         * msk : 0.f;

    const int ln15 = pixel & 15;
    const int nb   = pixel >> 4;
#pragma unroll 4
    for (int c = 0; c < 64; c += 2) {
        const float* p = feat + ((b * 64 + c) << 12);
        const float* q = p + 4096;
        const float v0 = p[i00] * w00 + p[i01] * w01 + p[i10] * w10 + p[i11] * w11;
        const float v1 = q[i00] * w00 + q[i01] * w01 + q[i10] * w10 + q[i11] * w11;
        const int s  = tap * 2 + (c >> 5);
        const int kl = c & 31;                         // even
        const int gg = (kl >> 3) & 1;
        const int e  = (kl & 7) | ((kl >> 4) << 3);    // even; c+1 -> e+1
        const unsigned pair = (unsigned)f2bf(v0) | ((unsigned)f2bf(v1) << 16);
        *(unsigned*)&sp[(((s * 2048 + nb) * 32 + gg * 16 + ln15) << 4) + e] = pair;
    }
}

// ---------------------------------------------------------------------------
// Kernel 3: main GEMM  out[64][NPIX] = weight[64x576] x sampled[576xNPIX].
// 256 blocks x 8 waves; each wave owns one 16-pixel N-tile and all 64 output
// channels.  A-fragments (73728 B) DMA'd into LDS once per block by the TDM,
// then read back as ds_load_b128.  B fragments software-pipelined one K-step
// ahead.
// ---------------------------------------------------------------------------
__global__ void __launch_bounds__(256) dcn_gemm_kernel(
    const unsigned short* __restrict__ a2, const unsigned short* __restrict__ sp,
    float* __restrict__ out)
{
    extern __shared__ int4 sA[];                 // 4608 int4 = 73728 bytes
    const int tid  = threadIdx.x;
    const int lane = tid & 31;
    const int wv   = tid >> 5;

    if (wv == 0) tdm_load_pack_to_lds0(a2);
    __syncthreads();

    const int nb = blockIdx.x * 8 + wv;          // 0..2047
    v8f acc[4] = {};
    const int4* bp = (const int4*)sp;

    BFrag bcur, bnxt;
    {
        const int boff = (nb * 32 + lane) * 2;   // s = 0
        bcur.i4[0] = bp[boff]; bcur.i4[1] = bp[boff + 1];
        bnxt = bcur;
    }
    for (int s = 0; s < 18; ++s) {
        if (s < 17) {                            // prefetch next B fragment
            const int bo = (((s + 1) * 2048 + nb) * 32 + lane) * 2;
            bnxt.i4[0] = bp[bo]; bnxt.i4[1] = bp[bo + 1];
        }
#pragma unroll
        for (int mt = 0; mt < 4; ++mt) {
            BFrag af;
            const int aoff = ((s * 4 + mt) * 32 + lane) * 2;
            af.i4[0] = sA[aoff];
            af.i4[1] = sA[aoff + 1];
            acc[mt] = wmma_bf16(af.v, bcur.v, acc[mt]);
        }
        bcur = bnxt;
    }

    const int g   = lane >> 4;
    const int pix = nb * 16 + (lane & 15);
    const int b   = pix >> 12;
    const int hw  = pix & 4095;
#pragma unroll
    for (int mt = 0; mt < 4; ++mt) {
#pragma unroll
        for (int r = 0; r < 8; ++r) {
            const int o = mt * 16 + g * 8 + r;   // output channel
            out[((b * 64 + o) << 12) + hw] = acc[mt][r];
        }
    }
}

// ---------------------------------------------------------------------------
// Launch.  Workspace map (bytes):
//   [0,        73728)   A1 fragment pack  (om_weight, bf16)
//   [73728,   147456)   A2 fragment pack  (weight, bf16)
//   [147456, 3686400)   om[27][NPIX]      (f32: offsets + sigmoid mask)
//   [3686400, 41435136) sampled B-frag pack (bf16, 576 x NPIX)
// Total ~39.5 MB.
// ---------------------------------------------------------------------------
extern "C" void kernel_launch(void* const* d_in, const int* in_sizes, int n_in,
                              void* d_out, int out_size, void* d_ws, size_t ws_size,
                              hipStream_t stream) {
    const float* input_feat = (const float*)d_in[0];
    const float* degrad     = (const float*)d_in[1];
    const float* weight     = (const float*)d_in[2];
    const float* om_weight  = (const float*)d_in[3];
    const float* om_bias    = (const float*)d_in[4];
    float* out = (float*)d_out;

    char* ws = (char*)d_ws;
    unsigned short* a1pack = (unsigned short*)(ws);
    unsigned short* a2pack = (unsigned short*)(ws + 73728);
    float*          om     = (float*)(ws + 147456);
    unsigned short* spack  = (unsigned short*)(ws + 3686400);

    pack_a1_kernel<<<9, 256, 0, stream>>>(om_weight, a1pack);
    pack_a2_kernel<<<9, 256, 0, stream>>>(weight, a2pack);
    om_conv_kernel<<<512, 128, 73728 + 15840, stream>>>(input_feat, degrad,
                                                        om_bias, a1pack, om);
    sample_pack_kernel<<<1152, 256, 0, stream>>>(input_feat, om, spack);
    dcn_gemm_kernel<<<256, 256, 73728, stream>>>(a2pack, spack, out);
}
